// KASPPS_4466765988471
// MI455X (gfx1250) — matrix operand order repaired and checked by reading, not compiled, not run on hardware
//
#include <hip/hip_runtime.h>
#include <hip/hip_bf16.h>

typedef __bf16 bf16_t;
typedef __attribute__((ext_vector_type(16))) __bf16 v16bf;
typedef __attribute__((ext_vector_type(8)))  float  v8f;

#define PIX   8192   // B*H*W = 2*64*64
#define HWPIX 4096   // H*W
#define LDA   72     // LDS row stride (bf16) with pad, 144B = 9*16B aligned
#define LDB   72

__device__ __forceinline__ float silu_f(float v) { return v / (1.f + __expf(-v)); }

// Async global->LDS copy of 16B per lane (CDNA5 GLOBAL_LOAD_ASYNC_TO_LDS_B128,
// GV mode, tracked by ASYNCcnt). ldsoff = LDS byte offset (= low 32 bits of a
// generic pointer to __shared__, per the flat-aperture rule).
__device__ __forceinline__ void async_copy_b128(unsigned int ldsoff,
                                                const void* gsrc)
{
    unsigned long long ga = (unsigned long long)(uintptr_t)gsrc;
    asm volatile("global_load_async_to_lds_b128 %0, %1, off"
                 :: "v"(ldsoff), "v"(ga) : "memory");
}
__device__ __forceinline__ void async_wait0()
{
    asm volatile("s_wait_asynccnt 0x0" ::: "memory");
}

// ---------------------------------------------------------------------------
// Per-pixel feature vector: f[p][i*(G+4)+c], c=0 -> silu(x), c=1.. -> bspline
// bases. p == PIX is the "zero pixel" row (value 0.0) used for OOB taps.
// ---------------------------------------------------------------------------
__global__ void precompute_f(const float* __restrict__ x, bf16_t* __restrict__ f,
                             int G, int C)
{
    int idx = blockIdx.x * blockDim.x + threadIdx.x;
    int total = (PIX + 1) * 64;
    if (idx >= total) return;
    int i = idx & 63;
    int p = idx >> 6;
    float v = 0.f;
    if (p < PIX) {
        int b = p >> 12;
        int hw = p & 4095;
        v = x[((long long)(b * 64 + i) << 12) + hw];
    }
    float si = silu_f(v);
    // Cox-de Boor, order 3, uniform grid t_j = -1 + (j-3)*h, j=0..G+6
    float hstep = 2.0f / (float)G;
    float bas[16];
    int n0 = G + 6;
    for (int j = 0; j < n0; ++j) {
        float t0 = -1.f + (float)(j - 3) * hstep;
        bas[j] = (v >= t0 && v < t0 + hstep) ? 1.f : 0.f;
    }
    for (int ord = 1; ord <= 3; ++ord) {
        int n = n0 - ord;
        float inv = 1.f / ((float)ord * hstep);
        for (int j = 0; j < n; ++j) {
            float tj = -1.f + (float)(j - 3) * hstep;
            float left  = (v - tj) * inv * bas[j];
            float right = ((tj + (float)(ord + 1) * hstep) - v) * inv * bas[j + 1];
            bas[j] = left + right;
        }
    }
    bf16_t* out = f + (long long)p * C + i * (G + 4);
    out[0] = (bf16_t)si;
    for (int g2 = 0; g2 < G + 3; ++g2) out[1 + g2] = (bf16_t)bas[g2];
}

// ---------------------------------------------------------------------------
// Transposed bf16 weights: WT[o][k*C + i*(G+4)+c]; c==0 base_w, else spline_w.
// ---------------------------------------------------------------------------
__global__ void prep_weights(const float* __restrict__ base_w,
                             const float* __restrict__ spline_w,
                             bf16_t* __restrict__ WT, int G, int C, int Ktot)
{
    int idx = blockIdx.x * blockDim.x + threadIdx.x;
    if (idx >= 64 * Ktot) return;
    int o = idx / Ktot;
    int r = idx % Ktot;
    int k = r / C;
    int rem = r % C;
    int i = rem / (G + 4);
    int c = rem % (G + 4);
    float w = (c == 0) ? base_w[(o * 64 + i) * 9 + k]
                       : spline_w[((o * 64 + i) * 9 + k) * (G + 3) + (c - 1)];
    WT[idx] = (bf16_t)w;
}

__global__ void prep_fuse_w(const float* __restrict__ fuse_w, bf16_t* __restrict__ WfT)
{
    int idx = blockIdx.x * blockDim.x + threadIdx.x;
    if (idx >= 64 * 192) return;
    int o = idx / 192, k = idx % 192;
    WfT[idx] = (bf16_t)fuse_w[o * 256 + k];
}

// ---------------------------------------------------------------------------
// KAN branch GEMM, implicit im2col. Block = one image row (64 pixels) x 64 out
// channels. 8 waves: wave w -> M subtile (w&3), N half (w>>2), 2 accumulators.
// Tile staging uses CDNA5 async global->LDS (no VGPR round-trip).
// ---------------------------------------------------------------------------
__global__ __launch_bounds__(256)
void kan_gemm(const bf16_t* __restrict__ f, const bf16_t* __restrict__ WT,
              float* __restrict__ y, int C, int dil)
{
    __shared__ bf16_t As[64 * LDA];
    __shared__ bf16_t Bs[64 * LDB];
    int tid = threadIdx.x;
    int lane = tid & 31, wave = tid >> 5;
    int mt = wave & 3, nt = wave >> 2;
    int pix0 = blockIdx.x * 64;           // one row of the image: fixed (b,h)
    int b = pix0 >> 12;
    int h = (pix0 >> 6) & 63;
    int Ktot = 9 * C;
    v8f acc0 = {}; v8f acc1 = {};
    int m = lane & 15, half = lane >> 4;

    // per-thread staging coordinates: row r, 16-element column group
    int r = tid >> 2;
    int col = (tid & 3) * 16;
    unsigned int ldsA = (unsigned int)(uintptr_t)&As[r * LDA + col];
    unsigned int ldsB = (unsigned int)(uintptr_t)&Bs[r * LDB + col];

    for (int tap = 0; tap < 9; ++tap) {
        int dh = (tap / 3 - 1) * dil;
        int dw = (tap % 3 - 1) * dil;
        int hs = h + dh;
        bool hok = (hs >= 0 && hs < 64);
        int wsrc = r + dw;
        long long srcp = (hok && wsrc >= 0 && wsrc < 64)
                             ? (long long)(b * 4096 + hs * 64 + wsrc)
                             : (long long)PIX; // zero-pixel basis row
        const bf16_t* srcA = f + srcp * C + col;
        const bf16_t* srcB = WT + (long long)r * Ktot + tap * C + col;
        for (int cb = 0; cb < C; cb += 64) {
            // A tile: 64 pixels x 64 K (bf16), bounds-checked tap shift
            async_copy_b128(ldsA,      srcA + cb);
            async_copy_b128(ldsA + 16, srcA + cb + 8);
            // B tile: Bs[n][kk] from WT[n][tap*C + cb + kk]
            async_copy_b128(ldsB,      srcB + cb);
            async_copy_b128(ldsB + 16, srcB + cb + 8);
            async_wait0();
            __syncthreads();
            #pragma unroll
            for (int kk0 = 0; kk0 < 64; kk0 += 32) {
                v16bf a;
                const bf16_t* Ar = &As[(mt * 16 + m) * LDA + kk0];
                #pragma unroll
                for (int vv = 0; vv < 8; ++vv) {
                    int k = half * 16 + 2 * vv;
                    a[2 * vv]     = Ar[k];
                    a[2 * vv + 1] = Ar[k + 1];
                }
                v16bf b0, b1;
                const bf16_t* Br0 = &Bs[(nt * 32 + m) * LDB + kk0 + half * 16];
                const bf16_t* Br1 = &Bs[(nt * 32 + 16 + m) * LDB + kk0 + half * 16];
                #pragma unroll
                for (int e = 0; e < 16; ++e) { b0[e] = Br0[e]; b1[e] = Br1[e]; }
                acc0 = __builtin_amdgcn_wmma_f32_16x16x32_bf16(false, a, false, b0,
                                                               (short)0, acc0, false, false);
                acc1 = __builtin_amdgcn_wmma_f32_16x16x32_bf16(false, a, false, b1,
                                                               (short)0, acc1, false, false);
            }
            __syncthreads();
        }
    }
    #pragma unroll
    for (int rr = 0; rr < 8; ++rr) {
        int mrow = mt * 16 + rr + 8 * half;
        long long base = (long long)(pix0 + mrow) * 64;
        y[base + nt * 32 + m]      = acc0[rr];
        y[base + nt * 32 + 16 + m] = acc1[rr];
    }
}

// ---------------------------------------------------------------------------
// BN stats -> affine:  out = y*A + B  (A = g*rsqrt(var+eps), B = b - mean*A)
// ---------------------------------------------------------------------------
__global__ void bn_stats(const float* __restrict__ y, const float* __restrict__ g,
                         const float* __restrict__ bb, float* __restrict__ A,
                         float* __restrict__ B, int npix)
{
    int c = blockIdx.x;
    float s = 0.f, s2 = 0.f;
    for (int p = threadIdx.x; p < npix; p += blockDim.x) {
        float v = y[(long long)p * 64 + c];
        s += v; s2 += v * v;
    }
    __shared__ float sh[256], sh2[256];
    sh[threadIdx.x] = s; sh2[threadIdx.x] = s2;
    __syncthreads();
    for (int st = 128; st > 0; st >>= 1) {
        if (threadIdx.x < st) { sh[threadIdx.x] += sh[threadIdx.x + st];
                                sh2[threadIdx.x] += sh2[threadIdx.x + st]; }
        __syncthreads();
    }
    if (threadIdx.x == 0) {
        float mean = sh[0] / (float)npix;
        float var  = sh2[0] / (float)npix - mean * mean;
        float a = g[c] * rsqrtf(var + 1e-5f);
        A[c] = a; B[c] = bb[c] - mean * a;
    }
}

__global__ void se_reduce(const float* __restrict__ y, const float* __restrict__ A,
                          const float* __restrict__ B, float* __restrict__ semean)
{
    int b = blockIdx.x >> 6, c = blockIdx.x & 63;
    float s = 0.f;
    for (int p = threadIdx.x; p < HWPIX; p += blockDim.x) {
        float v = y[(long long)(b * HWPIX + p) * 64 + c] * A[c] + B[c];
        s += fmaxf(v, 0.f);
    }
    __shared__ float sh[256];
    sh[threadIdx.x] = s; __syncthreads();
    for (int st = 128; st > 0; st >>= 1) {
        if (threadIdx.x < st) sh[threadIdx.x] += sh[threadIdx.x + st];
        __syncthreads();
    }
    if (threadIdx.x == 0) semean[b * 64 + c] = sh[0] / (float)HWPIX;
}

__global__ void se_mlp(const float* __restrict__ semean, const float* __restrict__ w1,
                       const float* __restrict__ w2, float* __restrict__ ses)
{
    int b = blockIdx.x, c = threadIdx.x;
    __shared__ float sm[64], t1[4];
    sm[c] = semean[b * 64 + c];
    __syncthreads();
    if (c < 4) {
        float s = 0.f;
        for (int i = 0; i < 64; ++i) s += sm[i] * w1[c * 64 + i];
        t1[c] = fmaxf(s, 0.f);
    }
    __syncthreads();
    float s = 0.f;
    for (int r = 0; r < 4; ++r) s += t1[r] * w2[c * 4 + r];
    ses[b * 64 + c] = 1.f / (1.f + __expf(-s));
}

// ------------------------- global-pool branch ------------------------------
__global__ void gp_pool(const float* __restrict__ x, float* __restrict__ xmean)
{
    int bi = blockIdx.x; // b*64 + i
    float s = 0.f;
    for (int p = threadIdx.x; p < HWPIX; p += blockDim.x)
        s += x[((long long)bi << 12) + p];
    __shared__ float sh[256];
    sh[threadIdx.x] = s; __syncthreads();
    for (int st = 128; st > 0; st >>= 1) {
        if (threadIdx.x < st) sh[threadIdx.x] += sh[threadIdx.x + st];
        __syncthreads();
    }
    if (threadIdx.x == 0) xmean[bi] = sh[0] / (float)HWPIX;
}

// one block, 64 threads: 1x1 conv + BN(over batch) + ReLU + SE + fuse constant
__global__ void gp_small(const float* __restrict__ xmean,
                         const float* __restrict__ cw, const float* __restrict__ cb,
                         const float* __restrict__ g,  const float* __restrict__ bb,
                         const float* __restrict__ w1, const float* __restrict__ w2,
                         const float* __restrict__ fuse_w, const float* __restrict__ fuse_b,
                         float* __restrict__ cadd)
{
    int c = threadIdx.x;
    __shared__ float rr[2][64], t1s[2][4], gse[2][64];
    float gv[2];
    for (int b = 0; b < 2; ++b) {
        float s = cb[c];
        for (int i = 0; i < 64; ++i) s += xmean[b * 64 + i] * cw[c * 64 + i];
        gv[b] = s;
    }
    float mean = 0.5f * (gv[0] + gv[1]);
    float var  = 0.5f * (gv[0] * gv[0] + gv[1] * gv[1]) - mean * mean;
    float a = g[c] * rsqrtf(var + 1e-5f);
    float bo = bb[c] - mean * a;
    rr[0][c] = fmaxf(gv[0] * a + bo, 0.f);
    rr[1][c] = fmaxf(gv[1] * a + bo, 0.f);
    __syncthreads();
    if (c < 8) {
        int b = c >> 2, r = c & 3;
        float s = 0.f;
        for (int i = 0; i < 64; ++i) s += rr[b][i] * w1[r * 64 + i];
        t1s[b][r] = fmaxf(s, 0.f);
    }
    __syncthreads();
    for (int b = 0; b < 2; ++b) {
        float s = 0.f;
        for (int r = 0; r < 4; ++r) s += t1s[b][r] * w2[c * 4 + r];
        gse[b][c] = rr[b][c] * (1.f / (1.f + __expf(-s)));
    }
    __syncthreads();
    for (int b = 0; b < 2; ++b) {
        float s = fuse_b[c];
        for (int i = 0; i < 64; ++i) s += fuse_w[c * 256 + 192 + i] * gse[b][i];
        cadd[b * 64 + c] = s;
    }
}

// ---------------------------------------------------------------------------
// Fuse 1x1 conv as WMMA GEMM: K = 192 (3 branches), BN/ReLU/SE applied while
// building A tiles; gp branch + fuse bias folded in as per-(b,o) constant.
// B tile staged with async global->LDS.
// ---------------------------------------------------------------------------
__global__ __launch_bounds__(256)
void fuse_gemm(const float* __restrict__ y0, const float* __restrict__ y1,
               const float* __restrict__ y2, const float* __restrict__ Ac,
               const float* __restrict__ Bc, const float* __restrict__ ses,
               const bf16_t* __restrict__ WfT, const float* __restrict__ cadd,
               float* __restrict__ z)
{
    __shared__ bf16_t As[64 * LDA];
    __shared__ bf16_t Bs[64 * LDB];
    int tid = threadIdx.x;
    int lane = tid & 31, wave = tid >> 5;
    int mt = wave & 3, nt = wave >> 2;
    int pix0 = blockIdx.x * 64;
    int bidx = pix0 >> 12;
    v8f acc0 = {}; v8f acc1 = {};
    int m = lane & 15, half = lane >> 4;

    int r = tid >> 2;
    int col = (tid & 3) * 16;
    unsigned int ldsB = (unsigned int)(uintptr_t)&Bs[r * LDB + col];
    const bf16_t* srcB = WfT + r * 192 + col;

    for (int cb = 0; cb < 192; cb += 64) {
        int br = cb >> 6;
        const float* yb = (br == 0) ? y0 : (br == 1) ? y1 : y2;
        // B tile via async copy
        async_copy_b128(ldsB,      srcB + cb);
        async_copy_b128(ldsB + 16, srcB + cb + 8);
        { // A tile with fused BN+ReLU+SE, fp32 -> bf16
            long long pix = pix0 + r;
            const float* src = yb + pix * 64 + col;
            bf16_t* dst = &As[r * LDA + col];
            #pragma unroll
            for (int j = 0; j < 16; ++j) {
                int c = col + j;
                float v = fmaxf(src[j] * Ac[br * 64 + c] + Bc[br * 64 + c], 0.f);
                dst[j] = (bf16_t)(v * ses[br * 128 + bidx * 64 + c]);
            }
        }
        async_wait0();
        __syncthreads();
        #pragma unroll
        for (int kk0 = 0; kk0 < 64; kk0 += 32) {
            v16bf a;
            const bf16_t* Ar = &As[(mt * 16 + m) * LDA + kk0];
            #pragma unroll
            for (int vv = 0; vv < 8; ++vv) {
                int k = half * 16 + 2 * vv;
                a[2 * vv]     = Ar[k];
                a[2 * vv + 1] = Ar[k + 1];
            }
            v16bf b0, b1;
            const bf16_t* Br0 = &Bs[(nt * 32 + m) * LDB + kk0 + half * 16];
            const bf16_t* Br1 = &Bs[(nt * 32 + 16 + m) * LDB + kk0 + half * 16];
            #pragma unroll
            for (int e = 0; e < 16; ++e) { b0[e] = Br0[e]; b1[e] = Br1[e]; }
            acc0 = __builtin_amdgcn_wmma_f32_16x16x32_bf16(false, a, false, b0,
                                                           (short)0, acc0, false, false);
            acc1 = __builtin_amdgcn_wmma_f32_16x16x32_bf16(false, a, false, b1,
                                                           (short)0, acc1, false, false);
        }
        __syncthreads();
    }
    #pragma unroll
    for (int rr = 0; rr < 8; ++rr) {
        int mrow = mt * 16 + rr + 8 * half;
        long long base = (long long)(pix0 + mrow) * 64;
        int o0 = nt * 32 + m;
        z[base + o0]      = acc0[rr] + cadd[bidx * 64 + o0];
        z[base + o0 + 16] = acc1[rr] + cadd[bidx * 64 + o0 + 16];
    }
}

// Final BN+ReLU and transpose to NCHW output layout.
__global__ void finalize(const float* __restrict__ z, const float* __restrict__ A,
                         const float* __restrict__ B, float* __restrict__ out)
{
    int b = blockIdx.x >> 6, o = blockIdx.x & 63;
    for (int p = threadIdx.x; p < HWPIX; p += blockDim.x) {
        float v = z[(long long)(b * HWPIX + p) * 64 + o] * A[o] + B[o];
        out[((long long)(b * 64 + o) << 12) + p] = fmaxf(v, 0.f);
    }
}

extern "C" void kernel_launch(void* const* d_in, const int* in_sizes, int n_in,
                              void* d_out, int out_size, void* d_ws, size_t ws_size,
                              hipStream_t stream)
{
    const float* x = (const float*)d_in[0];
    const float* base_w[3]  = {(const float*)d_in[1],  (const float*)d_in[7],  (const float*)d_in[13]};
    const float* spline_w[3]= {(const float*)d_in[2],  (const float*)d_in[8],  (const float*)d_in[14]};
    const float* bn_g[3]    = {(const float*)d_in[3],  (const float*)d_in[9],  (const float*)d_in[15]};
    const float* bn_b[3]    = {(const float*)d_in[4],  (const float*)d_in[10], (const float*)d_in[16]};
    const float* se_w1[3]   = {(const float*)d_in[5],  (const float*)d_in[11], (const float*)d_in[17]};
    const float* se_w2[3]   = {(const float*)d_in[6],  (const float*)d_in[12], (const float*)d_in[18]};
    const float* gp_conv_w  = (const float*)d_in[19];
    const float* gp_conv_b  = (const float*)d_in[20];
    const float* gp_bn_g    = (const float*)d_in[21];
    const float* gp_bn_b    = (const float*)d_in[22];
    const float* gp_se_w1   = (const float*)d_in[23];
    const float* gp_se_w2   = (const float*)d_in[24];
    const float* fuse_w     = (const float*)d_in[25];
    const float* fuse_b     = (const float*)d_in[26];
    const float* fuse_bn_g  = (const float*)d_in[27];
    const float* fuse_bn_b  = (const float*)d_in[28];

    const int Gs[3]   = {3, 6, 9};
    const int dils[3] = {6, 12, 18};

    char* wsp = (char*)d_ws;
    size_t off = 0;
    auto alloc = [&](size_t bytes) -> void* {
        void* p = wsp + off;
        off = (off + bytes + 255) & ~(size_t)255;
        return p;
    };

    bf16_t* f[3]; bf16_t* WT[3]; float* ybuf[3];
    for (int br = 0; br < 3; ++br) {
        int C = 64 * (Gs[br] + 4);
        f[br]    = (bf16_t*)alloc((size_t)(PIX + 1) * C * 2);
        WT[br]   = (bf16_t*)alloc((size_t)64 * 9 * C * 2);
        ybuf[br] = (float*)alloc((size_t)PIX * 64 * 4);
    }
    float*  zbuf  = (float*)alloc((size_t)PIX * 64 * 4);
    float*  Ac    = (float*)alloc(3 * 64 * 4);
    float*  Bc    = (float*)alloc(3 * 64 * 4);
    float*  Af    = (float*)alloc(64 * 4);
    float*  Bf    = (float*)alloc(64 * 4);
    float*  semean= (float*)alloc(3 * 128 * 4);
    float*  ses   = (float*)alloc(3 * 128 * 4);
    float*  xmean = (float*)alloc(128 * 4);
    float*  cadd  = (float*)alloc(128 * 4);
    bf16_t* WfT   = (bf16_t*)alloc(64 * 192 * 2);

    // Phase 1: feature + weight prep
    for (int br = 0; br < 3; ++br) {
        int C = 64 * (Gs[br] + 4), Ktot = 9 * C;
        int tpre = (PIX + 1) * 64;
        precompute_f<<<(tpre + 255) / 256, 256, 0, stream>>>(x, f[br], Gs[br], C);
        int twt = 64 * Ktot;
        prep_weights<<<(twt + 255) / 256, 256, 0, stream>>>(base_w[br], spline_w[br],
                                                            WT[br], Gs[br], C, Ktot);
    }
    prep_fuse_w<<<48, 256, 0, stream>>>(fuse_w, WfT);

    // Phase 2: branch GEMMs (WMMA + async-LDS staging) + BN/SE stats
    for (int br = 0; br < 3; ++br) {
        int C = 64 * (Gs[br] + 4);
        kan_gemm<<<128, 256, 0, stream>>>(f[br], WT[br], ybuf[br], C, dils[br]);
        bn_stats<<<64, 256, 0, stream>>>(ybuf[br], bn_g[br], bn_b[br],
                                         Ac + br * 64, Bc + br * 64, PIX);
        se_reduce<<<128, 256, 0, stream>>>(ybuf[br], Ac + br * 64, Bc + br * 64,
                                           semean + br * 128);
        se_mlp<<<2, 64, 0, stream>>>(semean + br * 128, se_w1[br], se_w2[br],
                                     ses + br * 128);
    }

    // Phase 3: global-pool branch -> per-(b,o) fuse constant
    gp_pool<<<128, 256, 0, stream>>>(x, xmean);
    gp_small<<<1, 64, 0, stream>>>(xmean, gp_conv_w, gp_conv_b, gp_bn_g, gp_bn_b,
                                   gp_se_w1, gp_se_w2, fuse_w, fuse_b, cadd);

    // Phase 4: fuse GEMM (WMMA) + final BN/ReLU
    fuse_gemm<<<128, 256, 0, stream>>>(ybuf[0], ybuf[1], ybuf[2], Ac, Bc, ses,
                                       WfT, cadd, zbuf);
    bn_stats<<<64, 256, 0, stream>>>(zbuf, fuse_bn_g, fuse_bn_b, Af, Bf, PIX);
    finalize<<<128, 256, 0, stream>>>(zbuf, Af, Bf, (float*)d_out);
}
